// Root_Integ_16097537426000
// MI455X (gfx1250) — matrix operand order repaired and checked by reading, not compiled, not run on hardware
//
#include <hip/hip_runtime.h>
#include <math.h>

typedef __attribute__((ext_vector_type(16))) _Float16 v16h;
typedef __attribute__((ext_vector_type(8)))  float    v8f;

#define SUB      64
#define TT       384                 // time-tile per block
#define HALO     200                 // T_hist (fixed by reference)
#define ZC_TILES 37                  // ceil((TT+HALO)/16)
#define ZC_ROWS  (ZC_TILES * 16)     // 592
#define RAW_TILES (TT / 16)          // 24

#define ZC_FLOATS   ((ZC_ROWS + 1) * SUB)   // +1 guard row in front
#define RAW_FLOATS  (TT * SUB)
#define KERN_FLOATS (HALO * SUB)
#define SH_FLOATS   (ZC_FLOATS + RAW_FLOATS + KERN_FLOATS)

#define WMMA_F16(a, b, c) \
    __builtin_amdgcn_wmma_f32_16x16x32_f16(false, (a), false, (b), (short)0, (c), false, false)

// ---- fragment loaders -------------------------------------------------------
// A (16x32 f16) per-lane layout: row M = lane%16;
//   lanes 0-15 : halves 0-7 = K0..7,  halves 8-15 = K16..23
//   lanes16-31 : halves 0-7 = K8..15, halves 8-15 = K24..31
// caller passes koff = kchunk + (lane<16 ? 0 : 8); row r already includes lane%16.
__device__ __forceinline__ v16h load_a_cvt(const float* __restrict__ src, long r, int T, int koff) {
    v16h a = {};
    if (r >= 0 && r < T) {
        const float4* p = reinterpret_cast<const float4*>(src + r * SUB + koff);
        float4 q0 = p[0], q1 = p[1];   // K koff..koff+7
        float4 q2 = p[4], q3 = p[5];   // K koff+16..koff+23
        a[0]=(_Float16)q0.x; a[1]=(_Float16)q0.y; a[2]=(_Float16)q0.z; a[3]=(_Float16)q0.w;
        a[4]=(_Float16)q1.x; a[5]=(_Float16)q1.y; a[6]=(_Float16)q1.z; a[7]=(_Float16)q1.w;
        a[8]=(_Float16)q2.x; a[9]=(_Float16)q2.y; a[10]=(_Float16)q2.z; a[11]=(_Float16)q2.w;
        a[12]=(_Float16)q3.x; a[13]=(_Float16)q3.y; a[14]=(_Float16)q3.z; a[15]=(_Float16)q3.w;
    }
    return a;
}

#define CVTHL(i, v) { float _v=(v); _Float16 _h=(_Float16)_v; hi[i]=_h; lo[i]=(_Float16)(_v-(float)_h); }

__device__ __forceinline__ void load_a_hilo(const float* __restrict__ src, long r, int T, int koff,
                                            v16h& hi, v16h& lo) {
    v16h z = {};
    hi = z; lo = z;
    if (r >= 0 && r < T) {
        const float4* p = reinterpret_cast<const float4*>(src + r * SUB + koff);
        float4 q0 = p[0], q1 = p[1], q2 = p[4], q3 = p[5];
        CVTHL(0,  q0.x) CVTHL(1,  q0.y) CVTHL(2,  q0.z) CVTHL(3,  q0.w)
        CVTHL(4,  q1.x) CVTHL(5,  q1.y) CVTHL(6,  q1.z) CVTHL(7,  q1.w)
        CVTHL(8,  q2.x) CVTHL(9,  q2.y) CVTHL(10, q2.z) CVTHL(11, q2.w)
        CVTHL(12, q3.x) CVTHL(13, q3.y) CVTHL(14, q3.z) CVTHL(15, q3.w)
    }
}

// B (32x16 f16) per-lane layout (same K order as A): col N = lane%16;
//   lanes 0-15: halves 0-7 = K0..7, 8-15 = K16..23 ; lanes 16-31: halves 0-7 = K8..15, 8-15 = K24..31
// B = C_denᵀ, so lane reads row n of C_den, 8 floats at kbase and 8 floats at kbase+16.
__device__ __forceinline__ v16h load_b_frag(const float* __restrict__ C, int n, int kbase) {
    const float4* p = reinterpret_cast<const float4*>(C + n * SUB + kbase);
    float4 q0 = p[0], q1 = p[1], q2 = p[4], q3 = p[5];
    v16h b;
    b[0]=(_Float16)q0.x; b[1]=(_Float16)q0.y; b[2]=(_Float16)q0.z; b[3]=(_Float16)q0.w;
    b[4]=(_Float16)q1.x; b[5]=(_Float16)q1.y; b[6]=(_Float16)q1.z; b[7]=(_Float16)q1.w;
    b[8]=(_Float16)q2.x; b[9]=(_Float16)q2.y; b[10]=(_Float16)q2.z; b[11]=(_Float16)q2.w;
    b[12]=(_Float16)q3.x; b[13]=(_Float16)q3.y; b[14]=(_Float16)q3.z; b[15]=(_Float16)q3.w;
    return b;
}

// C/D f32 16x16: VGPR r -> (M = r + (lane<16?0:8), N = lane%16)
__device__ __forceinline__ void store_acc_lds(float* __restrict__ dst, int m0, int n0, int lane, v8f c) {
    int col = n0 + (lane & 15);
    int row = m0 + ((lane < 16) ? 0 : 8);
    #pragma unroll
    for (int r = 0; r < 8; ++r) dst[(row + r) * SUB + col] = c[r];
}

// ---- fused kernel -----------------------------------------------------------
__global__ __launch_bounds__(256)
void root_integ_fused(const float* __restrict__ S, const float* __restrict__ Y,
                      const float* __restrict__ Z, const float* __restrict__ Cden,
                      const float* __restrict__ Vo, const float* __restrict__ Wsub,
                      const float* __restrict__ theta, const float* __restrict__ Ksp,
                      const float* __restrict__ tau, const float* __restrict__ delta,
                      float* __restrict__ out, int T)
{
    extern __shared__ float sh[];
    float* zc      = sh + SUB;                 // row 0 == global row (t0 - HALO); sh[0..63] = guard row
    float* sh_raw  = sh + ZC_FLOATS;           // [TT][SUB]
    float* sh_kern = sh_raw + RAW_FLOATS;      // transposed: [k][s]

    const int tid  = threadIdx.x;
    const int lane = tid & 31;
    const int wave = tid >> 5;
    const int t0   = blockIdx.x * TT;

    // ---------- phase 1: synaptic kernel kern[s,k] -> sh_kern[k*64+s] ----------
    {
        const float it0 = __expf(-tau[0]);
        const float it1 = __expf(-tau[1]);
        const float it2 = __expf(-tau[2]);
        for (int e = tid; e < HALO * SUB; e += 256) {
            const int s = e & (SUB - 1);
            const int k = e >> 6;
            const float t  = fmaxf((float)k - delta[s], 0.0f);
            const float a0 = t * it0, a1 = t * it1, a2 = t * it2;
            sh_kern[e] = a0 * __expf(-a0) * Ksp[s * 3 + 0]
                       + a1 * __expf(-a1) * Ksp[s * 3 + 1]
                       + a2 * __expf(-a2) * Ksp[s * 3 + 2];
        }
    }

    // ---------- phase 2: WMMA matmuls  Zc = Z@Cᵀ (exact f16), raw = Y@Cᵀ (hi/lo split) ----------
    {
        const int n0   = (wave & 3) * 16;          // N-tile per wave
        const int grp  = wave >> 2;                // M-tile interleave group
        const int n    = n0 + (lane & 15);
        const int kbB  = (lane < 16) ? 0 : 8;
        const int koffA = (lane < 16) ? 0 : 8;
        const int mlane = lane & 15;

        const v16h b0 = load_b_frag(Cden, n, kbB);        // K 0..31
        const v16h b1 = load_b_frag(Cden, n, kbB + 32);   // K 32..63

        // Zc tile (with halo), rows [t0-HALO, t0-HALO+592)
        for (int i = grp; i < ZC_TILES; i += 2) {
            const long r = (long)t0 - HALO + i * 16 + mlane;
            v16h a0 = load_a_cvt(Z, r, T, 0 + koffA);
            v16h a1 = load_a_cvt(Z, r, T, 32 + koffA);
            v8f c = {};
            c = WMMA_F16(a0, b0, c);
            c = WMMA_F16(a1, b1, c);
            store_acc_lds(zc, i * 16, n0, lane, c);
        }
        // raw tile, rows [t0, t0+TT)
        for (int i = grp; i < RAW_TILES; i += 2) {
            const long r = (long)t0 + i * 16 + mlane;
            v16h h0, l0, h1, l1;
            load_a_hilo(Y, r, T, 0 + koffA, h0, l0);
            load_a_hilo(Y, r, T, 32 + koffA, h1, l1);
            v8f c = {};
            c = WMMA_F16(h0, b0, c);
            c = WMMA_F16(h1, b1, c);
            c = WMMA_F16(l0, b0, c);
            c = WMMA_F16(l1, b1, c);
            store_acc_lds(sh_raw, i * 16, n0, lane, c);
        }
    }
    __syncthreads();

    // ---------- phase 3: causal depthwise FIR (register sliding window) + pointwise ----------
    {
        const int s = tid & (SUB - 1);
        const int g = tid >> 6;                    // 0..3 -> quarter of the tile
        const float th_s = theta[s];
        const float W_s  = Wsub[s];
        const float vo   = Vo[0];

        for (int strip = 0; strip < (TT / 4) / 8; ++strip) {   // 12 strips of 8 outputs
            const int lt0 = g * (TT / 4) + strip * 8;          // local t of first output
            float acc[8], w[8];
            #pragma unroll
            for (int i = 0; i < 8; ++i) acc[i] = 0.0f;
            // window for k=0: output lt0+i needs zc row (lt0 + i + HALO - 1)
            #pragma unroll
            for (int i = 0; i < 8; ++i) w[i] = zc[(lt0 + i + HALO - 1) * SUB + s];

            #pragma unroll 8
            for (int k = 0; k < HALO; ++k) {
                const float kv = sh_kern[k * SUB + s];
                #pragma unroll
                for (int i = 0; i < 8; ++i) acc[i] = fmaf(kv, w[i], acc[i]);
                #pragma unroll
                for (int i = 7; i > 0; --i) w[i] = w[i - 1];
                // k=199 touches row lt0-1: lands in the guard row, value unused
                w[0] = zc[(lt0 + HALO - 2 - k) * SUB + s];
            }

            #pragma unroll
            for (int i = 0; i < 8; ++i) {
                const int t = t0 + lt0 + i;
                if (t < T) {
                    const float x = S[(long)t * SUB + s] + th_s + acc[i]
                                  + sh_raw[(lt0 + i) * SUB + s];
                    const float sig = 1.0f / (1.0f + __expf(-x));
                    out[(long)t * SUB + s] = fmaf(sig, W_s, vo);
                }
            }
        }
    }
}

// ---- host launch ------------------------------------------------------------
extern "C" void kernel_launch(void* const* d_in, const int* in_sizes, int n_in,
                              void* d_out, int out_size, void* d_ws, size_t ws_size,
                              hipStream_t stream) {
    const float* S     = (const float*)d_in[0];
    const float* Y     = (const float*)d_in[1];
    const float* Z     = (const float*)d_in[2];
    const float* Cden  = (const float*)d_in[3];
    const float* Vo    = (const float*)d_in[4];
    const float* Wsub  = (const float*)d_in[5];
    const float* theta = (const float*)d_in[6];
    const float* Ksp   = (const float*)d_in[7];
    const float* tau   = (const float*)d_in[8];
    const float* delta = (const float*)d_in[9];
    float* out = (float*)d_out;

    const int T  = in_sizes[0] / SUB;
    const int nb = (T + TT - 1) / TT;
    const size_t shbytes = (size_t)SH_FLOATS * sizeof(float);   // 301,312 B < 320 KB

    root_integ_fused<<<dim3(nb), dim3(256), shbytes, stream>>>(
        S, Y, Z, Cden, Vo, Wsub, theta, Ksp, tau, delta, out, T);
}
